// SelfAttention_26740466385520
// MI455X (gfx1250) — matrix operand order repaired and checked by reading, not compiled, not run on hardware
//
#include <hip/hip_runtime.h>
#include <hip/hip_bf16.h>

typedef __attribute__((ext_vector_type(16))) __bf16 v16bf;
typedef __attribute__((ext_vector_type(8)))  __bf16 v8bf;
typedef __attribute__((ext_vector_type(4)))  __bf16 v4bf;
typedef __attribute__((ext_vector_type(8)))  float  v8f;
typedef __attribute__((ext_vector_type(4)))  float  v4f;

// ---------------------------------------------------------------------------
// gfx1250 async global->LDS copy (16B per lane) + ASYNCcnt wait.
// The clang builtin's signature is toolchain-dependent, so emit the CDNA5 ISA
// directly: VDST = per-lane LDS byte offset, VADDR = 64-bit global address
// (ISA 15.18.3, tracked by ASYNCcnt).
// ---------------------------------------------------------------------------
__device__ __forceinline__ void async_copy_b128(const __bf16* gsrc, __bf16* ldst) {
  unsigned loff = (unsigned)(unsigned long long)
      (__attribute__((address_space(3))) void*)ldst;
  asm volatile("global_load_async_to_lds_b128 %0, %1, off"
               :: "v"(loff), "v"(gsrc) : "memory");
}

__device__ __forceinline__ void wait_async0() {
#if __has_builtin(__builtin_amdgcn_s_wait_asynccnt)
  __builtin_amdgcn_s_wait_asynccnt(0);
#else
  asm volatile("s_wait_asynccnt 0x0" ::: "memory");
#endif
}

// ---------------------------------------------------------------------------
// f32 -> bf16 conversion, 4 elements per lane
// ---------------------------------------------------------------------------
__global__ __launch_bounds__(256) void cvt_bf16_x4(const float* __restrict__ in,
                                                   __bf16* __restrict__ out, int n) {
  int i = (blockIdx.x * 256 + threadIdx.x) * 4;
  if (i + 3 < n) {
    v4f f = *(const v4f*)(in + i);
    v4bf b;
    b.x = (__bf16)f.x; b.y = (__bf16)f.y; b.z = (__bf16)f.z; b.w = (__bf16)f.w;
    *(v4bf*)(out + i) = b;
  }
}

// ---------------------------------------------------------------------------
// Transpose (+convert) to bf16: out[c][r] = (bf16)in[r][c].
// in: [R][C] (f32 or bf16), out: [C][R] bf16. 32x32 LDS tiles, 256 threads.
// ---------------------------------------------------------------------------
template <typename TIN>
__global__ __launch_bounds__(256) void transpose_to_bf16(
    const TIN* __restrict__ in, __bf16* __restrict__ out, int R, int C,
    long long bsI, long long bsO) {
  __shared__ float t[32][33];
  const int r0 = blockIdx.y * 32, c0 = blockIdx.x * 32;
  const int tx = threadIdx.x & 31, ty = threadIdx.x >> 5;  // 32 x 8
  const TIN* ib = in + (long long)blockIdx.z * bsI;
  __bf16* ob = out + (long long)blockIdx.z * bsO;
#pragma unroll
  for (int i = 0; i < 4; ++i)
    t[ty + i * 8][tx] = (float)ib[(long long)(r0 + ty + i * 8) * C + c0 + tx];
  __syncthreads();
#pragma unroll
  for (int i = 0; i < 4; ++i)
    ob[(long long)(c0 + ty + i * 8) * R + r0 + tx] = (__bf16)t[tx][ty + i * 8];
}

// ---------------------------------------------------------------------------
// Tiled WMMA GEMM with async double-buffered LDS staging:
//   C = scale * (A @ B^T) + bias, A: [M][K] bf16, B: [N][K] bf16.
// Block = 256 threads = 8 wave32s; tile 128x128x32; wave tile 64x32 (4x2 WMMA).
// causal != 0: skip blocks entirely above the diagonal.
// ---------------------------------------------------------------------------
template <bool OUT_BF16>
__global__ __launch_bounds__(256) void gemm_wmma(
    const __bf16* __restrict__ A, const __bf16* __restrict__ B,
    const float* __restrict__ bias, void* __restrict__ Cout,
    int M, int N, int K, int lda, int ldb, int ldc, float scale,
    long long bsA, long long bsB, long long bsC, int causal) {
  constexpr int BM = 128, BN = 128, BK = 32, LDW = BK + 8;

  __shared__ __align__(16) __bf16 As[2][BM][LDW];
  __shared__ __align__(16) __bf16 Bs[2][BN][LDW];

  const int tid  = threadIdx.x;
  const int lane = tid & 31;
  const int wave = tid >> 5;
  const int half = lane >> 4;   // K-half selector (ISA 16-bit fragment layout)
  const int l16  = lane & 15;   // row (A) / col (B,C) within a 16x16 tile
  const int wr   = wave >> 2;   // 0..1 wave row
  const int wc   = wave & 3;    // 0..3 wave col

  const int m0 = blockIdx.x * BM;
  const int n0 = blockIdx.y * BN;
  if (causal && n0 > m0) return;  // tile fully above diagonal -> masked out

  const __bf16* Ab = A + (long long)blockIdx.z * bsA;
  const __bf16* Bb = B + (long long)blockIdx.z * bsB;

  // stage one 128x32 K-tile of A and B into LDS buffer `buf` (async DMA)
  auto stage = [&](int buf, int k0) {
#pragma unroll
    for (int i = 0; i < 2; ++i) {
      int c = tid + i * 256;
      int r = c >> 2, o = (c & 3) * 8;
      async_copy_b128(Ab + (long long)(m0 + r) * lda + k0 + o, &As[buf][r][o]);
      async_copy_b128(Bb + (long long)(n0 + r) * ldb + k0 + o, &Bs[buf][r][o]);
    }
  };

  v8f acc[4][2];
#pragma unroll
  for (int r = 0; r < 4; ++r)
#pragma unroll
    for (int c = 0; c < 2; ++c)
      acc[r][c] = (v8f){0.f, 0.f, 0.f, 0.f, 0.f, 0.f, 0.f, 0.f};

  const int KT = K / BK;
  stage(0, 0);
  int cur = 0;
  for (int kt = 0; kt < KT; ++kt) {
    wait_async0();      // this wave's async tile writes have landed in LDS
    __syncthreads();    // every wave's writes have landed; prior reads retired
    if (kt + 1 < KT) stage(cur ^ 1, (kt + 1) * BK);  // DMA next tile

    // ---- fragments per ISA 16-bit VGPR layouts (contiguous ds_load_b128) ----
    v16bf af[4], bfrag[2];
#pragma unroll
    for (int r = 0; r < 4; ++r) {
      const __bf16* ap = &As[cur][wr * 64 + r * 16 + l16][0];
      v8bf lo = *(const v8bf*)(ap + half * 8);       // K = half*8 + 0..7
      v8bf hi = *(const v8bf*)(ap + 16 + half * 8);  // K = 16 + half*8 + 0..7
      af[r] = __builtin_shufflevector(lo, hi, 0, 1, 2, 3, 4, 5, 6, 7,
                                      8, 9, 10, 11, 12, 13, 14, 15);
    }
#pragma unroll
    for (int c = 0; c < 2; ++c) {
      const __bf16* bp = &Bs[cur][wc * 32 + c * 16 + l16][0];
      v8bf lo = *(const v8bf*)(bp + half * 8);       // K = half*8 + 0..7
      v8bf hi = *(const v8bf*)(bp + 16 + half * 8);  // K = 16 + half*8 + 0..7
      bfrag[c] = __builtin_shufflevector(lo, hi, 0, 1, 2, 3, 4, 5, 6, 7,
                                         8, 9, 10, 11, 12, 13, 14, 15);
    }
    // ---- 8 WMMAs: 64x32 wave tile ----
#pragma unroll
    for (int r = 0; r < 4; ++r)
#pragma unroll
      for (int c = 0; c < 2; ++c)
        acc[r][c] = __builtin_amdgcn_wmma_f32_16x16x32_bf16(
            false, af[r], false, bfrag[c], (short)0, acc[r][c], false, false);
    cur ^= 1;
  }

  // ---- epilogue: C layout (lanes 0-15: M=v, N=lane; 16-31: M=8+v, N=lane-16)
#pragma unroll
  for (int r = 0; r < 4; ++r) {
    const int gm = m0 + wr * 64 + r * 16 + half * 8;
#pragma unroll
    for (int c = 0; c < 2; ++c) {
      const int gn = n0 + wc * 32 + c * 16 + l16;
      const float bv = bias ? bias[gn] : 0.0f;
#pragma unroll
      for (int v = 0; v < 8; ++v) {
        const float val = acc[r][c][v] * scale + bv;
        const long long idx =
            (long long)blockIdx.z * bsC + (long long)(gm + v) * ldc + gn;
        if constexpr (OUT_BF16)
          ((__bf16*)Cout)[idx] = (__bf16)val;
        else
          ((float*)Cout)[idx] = val;
      }
    }
  }
}

// ---------------------------------------------------------------------------
// Causal row softmax: reads f32 scores row prefix [0, row], writes bf16 P row
// (zero-filled above the diagonal so P@V runs as a plain GEMM).
// ---------------------------------------------------------------------------
__global__ __launch_bounds__(256) void softmax_causal(const float* __restrict__ sc,
                                                      __bf16* __restrict__ P, int S) {
  const int row = blockIdx.x, b = blockIdx.y, tid = threadIdx.x;
  const float* srow = sc + ((long long)b * S + row) * S;
  __bf16* prow = P + ((long long)b * S + row) * S;
  const int len = row + 1;
  __shared__ float red[256];

  float m = -3.0e38f;
  for (int j = tid; j < len; j += 256) m = fmaxf(m, srow[j]);
  red[tid] = m; __syncthreads();
  for (int s = 128; s > 0; s >>= 1) {
    if (tid < s) red[tid] = fmaxf(red[tid], red[tid + s]);
    __syncthreads();
  }
  m = red[0]; __syncthreads();

  float sum = 0.f;
  for (int j = tid; j < len; j += 256) sum += __expf(srow[j] - m);
  red[tid] = sum; __syncthreads();
  for (int s = 128; s > 0; s >>= 1) {
    if (tid < s) red[tid] += red[tid + s];
    __syncthreads();
  }
  const float inv = 1.0f / red[0];

  for (int j = tid; j < S; j += 256)
    prow[j] = (j < len) ? (__bf16)(__expf(srow[j] - m) * inv) : (__bf16)0.0f;
}

// ---------------------------------------------------------------------------
extern "C" void kernel_launch(void* const* d_in, const int* in_sizes, int n_in,
                              void* d_out, int out_size, void* d_ws, size_t ws_size,
                              hipStream_t stream) {
  (void)in_sizes; (void)n_in; (void)out_size; (void)ws_size;

  const float* x  = (const float*)d_in[0];
  const float* Wq = (const float*)d_in[1];
  const float* bq = (const float*)d_in[2];
  const float* Wk = (const float*)d_in[3];
  const float* bk = (const float*)d_in[4];
  const float* Wv = (const float*)d_in[5];
  const float* bv = (const float*)d_in[6];
  const float* Wo = (const float*)d_in[7];
  const float* bo = (const float*)d_in[8];
  float* out = (float*)d_out;

  const int Bn = 4, S = 2048, D = 1024;   // batch, seq, dim (KD=VD=OD=D)
  const int MS = Bn * S;                  // 8192 rows for fused-batch GEMMs

  // workspace layout (byte offsets, 200 MB total)
  char* ws = (char*)d_ws;
  const size_t MB = 1024u * 1024u;
  __bf16* xb  = (__bf16*)(ws + 0);         // 16 MB  x bf16
  __bf16* wqt = (__bf16*)(ws + 16 * MB);   //  2 MB  Wq^T bf16 [N][K]
  __bf16* wkt = (__bf16*)(ws + 18 * MB);   //  2 MB  Wk^T
  __bf16* wvt = (__bf16*)(ws + 20 * MB);   //  2 MB  Wv^T
  __bf16* wot = (__bf16*)(ws + 22 * MB);   //  2 MB  Wo^T
  __bf16* qb  = (__bf16*)(ws + 24 * MB);   // 16 MB
  __bf16* kb  = (__bf16*)(ws + 40 * MB);   // 16 MB
  __bf16* vb  = (__bf16*)(ws + 56 * MB);   // 16 MB
  __bf16* vt  = (__bf16*)(ws + 72 * MB);   // 16 MB  V^T per batch [D][S]
  float*  sc  = (float*) (ws + 88 * MB);   // 64 MB  scores f32 (L2-resident)
  __bf16* pb  = (__bf16*)(ws + 152 * MB);  // 32 MB  softmax probs bf16
  __bf16* yb  = (__bf16*)(ws + 184 * MB);  // 16 MB  attention output bf16

  const dim3 blk(256);
  const int nX = MS * D;

  // ---- inputs -> bf16 (weights transposed to the [N][K] operand layout) ----
  cvt_bf16_x4<<<dim3(nX / 4 / 256), 256, 0, stream>>>(x, xb, nX);
  transpose_to_bf16<float><<<dim3(D / 32, D / 32, 1), blk, 0, stream>>>(
      Wq, wqt, D, D, 0, 0);
  transpose_to_bf16<float><<<dim3(D / 32, D / 32, 1), blk, 0, stream>>>(
      Wk, wkt, D, D, 0, 0);
  transpose_to_bf16<float><<<dim3(D / 32, D / 32, 1), blk, 0, stream>>>(
      Wv, wvt, D, D, 0, 0);
  transpose_to_bf16<float><<<dim3(D / 32, D / 32, 1), blk, 0, stream>>>(
      Wo, wot, D, D, 0, 0);

  // ---- Q/K/V projections: [8192x1024] @ [1024x1024]^T + bias -> bf16 ----
  gemm_wmma<true><<<dim3(MS / 128, D / 128, 1), blk, 0, stream>>>(
      xb, wqt, bq, qb, MS, D, D, D, D, D, 1.0f, 0, 0, 0, 0);
  gemm_wmma<true><<<dim3(MS / 128, D / 128, 1), blk, 0, stream>>>(
      xb, wkt, bk, kb, MS, D, D, D, D, D, 1.0f, 0, 0, 0, 0);
  gemm_wmma<true><<<dim3(MS / 128, D / 128, 1), blk, 0, stream>>>(
      xb, wvt, bv, vb, MS, D, D, D, D, D, 1.0f, 0, 0, 0, 0);

  // ---- V^T per batch for the P@V operand ----
  transpose_to_bf16<__bf16><<<dim3(D / 32, S / 32, Bn), blk, 0, stream>>>(
      vb, vt, S, D, (long long)S * D, (long long)D * S);

  // ---- scores = (1/sqrt(D)) * Q @ K^T per batch (K is already [N][K]) ----
  gemm_wmma<false><<<dim3(S / 128, S / 128, Bn), blk, 0, stream>>>(
      qb, kb, nullptr, sc, S, S, D, D, D, S, 0.03125f,
      (long long)S * D, (long long)S * D, (long long)S * S, 1);

  // ---- causal softmax rows -> bf16 P ----
  softmax_causal<<<dim3(S, Bn), blk, 0, stream>>>(sc, pb, S);

  // ---- y = P @ V per batch (B operand = V^T [D][S]) ----
  gemm_wmma<true><<<dim3(S / 128, D / 128, Bn), blk, 0, stream>>>(
      pb, vt, nullptr, yb, S, D, S, S, S, D, 1.0f,
      (long long)S * S, (long long)D * S, (long long)S * D, 0);

  // ---- out = y @ Wo + bo -> f32 d_out ----
  gemm_wmma<false><<<dim3(MS / 128, D / 128, 1), blk, 0, stream>>>(
      yb, wot, bo, out, MS, D, D, D, D, D, 1.0f, 0, 0, 0, 0);
}
